// MultiheadAttention_16423954940642
// MI455X (gfx1250) — matrix operand order repaired and checked by reading, not compiled, not run on hardware
//
#include <hip/hip_runtime.h>
#include <hip/hip_bf16.h>
#include <math.h>

#define S_LEN   2048
#define NHEADS  16
#define HDIM    64
#define DMODEL  1024
#define BATCH   2
#define MROWS   (BATCH * S_LEN)   // 4096

typedef __attribute__((ext_vector_type(16))) _Float16 v16h;
typedef __attribute__((ext_vector_type(8)))  _Float16 v8h;
typedef __attribute__((ext_vector_type(8)))  float    v8f;

static __device__ __forceinline__ v8f wmma_f16(v16h a, v16h b, v8f c) {
  // D = A(16x32 f16) * B(32x16 f16) + C(16x16 f32)
  return __builtin_amdgcn_wmma_f32_16x16x32_f16(false, a, false, b, (short)0, c,
                                                false, false);
}

// A-fragment (16x32 f16): per lane, halves 0..7 = K(base..base+7),
// halves 8..15 = K(base+16..base+23), base = (lane<16 ? 0 : 8).
// p points at (row, base); reads p[0..7] and p[16..23].
static __device__ __forceinline__ v16h load_afrag(const _Float16* p) {
  v8h lo = *(const v8h*)p;
  v8h hi = *(const v8h*)(p + 16);
  v16h a;
#pragma unroll
  for (int i = 0; i < 8; ++i) { a[i] = lo[i]; a[i + 8] = hi[i]; }
  return a;
}

// B-fragment (32x16 f16): per lane, halves 0..7 = K(base..base+7),
// halves 8..15 = K(base+16..base+23), base = (lane<16 ? 0 : 8); col = lane&15.
// p points at (col, 16*(lane>>4)); reads from p - 8*(lane>>4).
static __device__ __forceinline__ v16h load_bfrag(const _Float16* p) {
  return load_afrag(p - ((threadIdx.x & 31) >> 4) * 8);
}

// Async copy of 16 bytes global -> LDS (per lane). Tracked by ASYNCcnt.
// LDS operand is the low 32 bits of the generic pointer (== DS byte address).
static __device__ __forceinline__ void async_copy16(_Float16* l,
                                                    const _Float16* g) {
  unsigned lo = (unsigned)(uintptr_t)l;
  asm volatile("global_load_async_to_lds_b128 %0, %1, off"
               :: "v"(lo), "v"(g) : "memory");
}

static __device__ __forceinline__ void wait_async0() {
  asm volatile("s_wait_asynccnt 0x0" ::: "memory");
}

// ---------------------------------------------------------------- converts
__global__ void k_f32_to_f16(const float* __restrict__ in,
                             _Float16* __restrict__ out, int n) {
  int i = blockIdx.x * blockDim.x + threadIdx.x;
  if (i < n) out[i] = (_Float16)in[i];
}

// out[c*R + r] = (f16) in[r*C + c]
__global__ void k_transpose_f32_to_f16(const float* __restrict__ in,
                                       _Float16* __restrict__ out, int R, int C) {
  int i = blockIdx.x * blockDim.x + threadIdx.x;
  if (i < R * C) {
    int r = i / C, c = i % C;
    out[(size_t)c * R + r] = (_Float16)in[(size_t)r * C + c];
  }
}

// ------------------------------------------------------- QKV projection
// grid = (MROWS/128, NHEADS, 3).  Block = 8 waves; wave tile 16 rows x 64 cols.
// W k-chunk (64n x 64k = 8KB) block-shared, async double-buffered in LDS.
// mode 0: Q(+RoPE,*1/8)->Qb[b,h,s,d]; 1: K(+RoPE)->Kb; 2: V->Vtb[b,h,d,s].
__global__ __launch_bounds__(256) void k_qkv_gemm(
    const _Float16* __restrict__ xh,
    const _Float16* __restrict__ Wqt, const _Float16* __restrict__ Wkt,
    const _Float16* __restrict__ Wvt,
    const float* __restrict__ bq, const float* __restrict__ bk,
    const float* __restrict__ bv,
    _Float16* __restrict__ Qb, _Float16* __restrict__ Kb,
    _Float16* __restrict__ Vtb) {
  __shared__ _Float16 Wl[2][64 * 72];   // 64 n-rows, stride 72 halves
  const int mode = blockIdx.z;
  const _Float16* W    = (mode == 0) ? Wqt : (mode == 1) ? Wkt : Wvt;
  const float*    bias = (mode == 0) ? bq : (mode == 1) ? bk : bv;
  const int h     = blockIdx.y;
  const int wave  = threadIdx.x >> 5;
  const int lane  = threadIdx.x & 31;
  const int ln    = lane & 15;
  const int hi    = lane >> 4;
  const int mrow0 = blockIdx.x * 128 + wave * 16;
  const int nb    = h * HDIM;

  // 512 x 16B chunks fill the 8KB tile; 2 chunks per thread
  const int c0 = threadIdx.x, c1 = threadIdx.x + 256;
  const int n0 = c0 >> 3, p0_ = (c0 & 7) * 8;
  const int n1 = c1 >> 3, p1_ = (c1 & 7) * 8;
  auto stage_w = [&](int buf, int kc) {
    async_copy16(&Wl[buf][n0 * 72 + p0_],
                 W + (size_t)(nb + n0) * DMODEL + kc + p0_);
    async_copy16(&Wl[buf][n1 * 72 + p1_],
                 W + (size_t)(nb + n1) * DMODEL + kc + p1_);
  };

  v8f acc[4] = {{}, {}, {}, {}};
  const _Float16* arowp = xh + (size_t)(mrow0 + ln) * DMODEL + hi * 8;

  stage_w(0, 0);
  v16h a0 = load_afrag(arowp);
  v16h a1 = load_afrag(arowp + 32);
  for (int kc = 0; kc < DMODEL; kc += 64) {
    const int cur = (kc >> 6) & 1;
    wait_async0();
    __syncthreads();
    if (kc + 64 < DMODEL) stage_w(cur ^ 1, kc + 64);
    v16h ac0 = a0, ac1 = a1;
    if (kc + 64 < DMODEL) {
      a0 = load_afrag(arowp + kc + 64);
      a1 = load_afrag(arowp + kc + 96);
    }
    const _Float16* Wb = &Wl[cur][0];
    v16h bf[8];
#pragma unroll
    for (int j = 0; j < 4; ++j) {
      bf[j * 2 + 0] = load_bfrag(Wb + (j * 16 + ln) * 72 + hi * 16);
      bf[j * 2 + 1] = load_bfrag(Wb + (j * 16 + ln) * 72 + 32 + hi * 16);
    }
#pragma unroll
    for (int j = 0; j < 4; ++j) acc[j] = wmma_f16(ac0, bf[j * 2 + 0], acc[j]);
#pragma unroll
    for (int j = 0; j < 4; ++j) acc[j] = wmma_f16(ac1, bf[j * 2 + 1], acc[j]);
  }

  if (mode < 2) {
    // RoPE: pair (d, d+32) = subtiles (j, j+2), same lane & VGPR slot.
#pragma unroll
    for (int j = 0; j < 2; ++j) {
#pragma unroll
      for (int r = 0; r < 8; ++r) {
        int grow = mrow0 + r + hi * 8;
        int bidx = grow >> 11;            // / S_LEN
        int spos = grow & (S_LEN - 1);
        int d1 = j * 16 + ln;
        float v1 = acc[j][r]     + bias[nb + d1];
        float v2 = acc[j + 2][r] + bias[nb + d1 + 32];
        float w = __expf(-(float)d1 * (9.210340371976184f / 32.0f));
        float sn, cs;
        __sincosf((float)spos * w, &sn, &cs);
        float o1 = v1 * cs - v2 * sn;
        float o2 = v2 * cs + v1 * sn;
        if (mode == 0) { o1 *= 0.125f; o2 *= 0.125f; }  // 1/sqrt(64)
        _Float16* dst = ((mode == 0) ? Qb : Kb) +
            ((size_t)(bidx * NHEADS + h) * S_LEN + spos) * HDIM;
        dst[d1]      = (_Float16)o1;
        dst[d1 + 32] = (_Float16)o2;
      }
    }
  } else {
#pragma unroll
    for (int j = 0; j < 4; ++j) {
#pragma unroll
      for (int r = 0; r < 8; ++r) {
        int grow = mrow0 + r + hi * 8;
        int bidx = grow >> 11;
        int spos = grow & (S_LEN - 1);
        int d = j * 16 + ln;
        float v = acc[j][r] + bias[nb + d];
        Vtb[((size_t)(bidx * NHEADS + h) * HDIM + d) * S_LEN + spos] = (_Float16)v;
      }
    }
  }
}

// ----------------------------------------------------- flash attention
// grid = (S_LEN/128, BATCH*NHEADS), block 256 (8 waves, 16 query rows each).
// 64-key chunks: K (64x64) and V (64x64, from Vt) async double-buffered.
// 16 WMMA per iteration, 32 iterations.
__global__ __launch_bounds__(256) void k_attention(
    const _Float16* __restrict__ Qb, const _Float16* __restrict__ Kb,
    const _Float16* __restrict__ Vtb, _Float16* __restrict__ Oh) {
  __shared__ _Float16 Kl[2][64 * 72];     // key rows, stride 72
  __shared__ _Float16 Vl[2][64 * 72];     // d rows, stride 72
  __shared__ _Float16 Plds[8][16 * 72];   // per-wave P tile (16 x 64)
  const int wave  = threadIdx.x >> 5;
  const int lane  = threadIdx.x & 31;
  const int ln    = lane & 15;
  const int hi    = lane >> 4;
  const int bh    = blockIdx.y;
  const int b     = bh >> 4;
  const int h     = bh & 15;
  const int qbase = blockIdx.x * 128 + wave * 16;
  const _Float16* Qp = Qb  + (size_t)bh * S_LEN * HDIM;
  const _Float16* Kp = Kb  + (size_t)bh * S_LEN * HDIM;
  const _Float16* Vp = Vtb + (size_t)bh * HDIM * S_LEN;
  _Float16* Pw = &Plds[wave][0];

  // 512 chunks per 8KB tile; 2 K-chunks + 2 V-chunks per thread
  const int c0 = threadIdx.x, c1 = threadIdx.x + 256;
  const int r0 = c0 >> 3, q0 = (c0 & 7) * 8;
  const int r1 = c1 >> 3, q1 = (c1 & 7) * 8;
  auto stage_kv = [&](int buf, int kb) {
    async_copy16(&Kl[buf][r0 * 72 + q0], Kp + (size_t)(kb + r0) * HDIM + q0);
    async_copy16(&Kl[buf][r1 * 72 + q1], Kp + (size_t)(kb + r1) * HDIM + q1);
    async_copy16(&Vl[buf][r0 * 72 + q0], Vp + (size_t)r0 * S_LEN + kb + q0);
    async_copy16(&Vl[buf][r1 * 72 + q1], Vp + (size_t)r1 * S_LEN + kb + q1);
  };

  const _Float16* qrow = Qp + (size_t)(qbase + ln) * HDIM + hi * 8;
  v16h aq0 = load_afrag(qrow);        // d = 0..31
  v16h aq1 = load_afrag(qrow + 32);   // d = 32..63

  float mrow[8], lrow[8];
#pragma unroll
  for (int r = 0; r < 8; ++r) { mrow[r] = -INFINITY; lrow[r] = 0.f; }
  v8f o[4] = {{}, {}, {}, {}};

  stage_kv(0, 0);
  for (int kb = 0; kb < S_LEN; kb += 64) {
    const int cur = (kb >> 6) & 1;
    wait_async0();
    __syncthreads();
    if (kb + 64 < S_LEN) stage_kv(cur ^ 1, kb + 64);

    const _Float16* Kt = &Kl[cur][0];
    const _Float16* Vt = &Vl[cur][0];

    // batched K B-fragments, then 8 score WMMAs
    v16h kf[8];
#pragma unroll
    for (int t = 0; t < 4; ++t) {
      kf[t * 2 + 0] = load_bfrag(Kt + (t * 16 + ln) * 72 + hi * 16);
      kf[t * 2 + 1] = load_bfrag(Kt + (t * 16 + ln) * 72 + 32 + hi * 16);
    }
    v8f s[4];
#pragma unroll
    for (int t = 0; t < 4; ++t) {
      v8f z = {};
      s[t] = wmma_f16(aq0, kf[t * 2 + 0], z);
    }
#pragma unroll
    for (int t = 0; t < 4; ++t) s[t] = wmma_f16(aq1, kf[t * 2 + 1], s[t]);

    // online softmax over 64 keys; C-fragment row r lives in a 16-lane half.
#pragma unroll
    for (int r = 0; r < 8; ++r) {
      float a0 = s[0][r], a1 = s[1][r], a2 = s[2][r], a3 = s[3][r];
      float mx = fmaxf(fmaxf(a0, a1), fmaxf(a2, a3));
#pragma unroll
      for (int off = 8; off >= 1; off >>= 1) mx = fmaxf(mx, __shfl_xor(mx, off, 16));
      float mn    = fmaxf(mrow[r], mx);
      float alpha = __expf(mrow[r] - mn);
      mrow[r] = mn;
      float p0 = __expf(a0 - mn);
      float p1 = __expf(a1 - mn);
      float p2 = __expf(a2 - mn);
      float p3 = __expf(a3 - mn);
      float rs = (p0 + p1) + (p2 + p3);
#pragma unroll
      for (int off = 8; off >= 1; off >>= 1) rs += __shfl_xor(rs, off, 16);
      lrow[r] = lrow[r] * alpha + rs;
      o[0][r] *= alpha; o[1][r] *= alpha; o[2][r] *= alpha; o[3][r] *= alpha;
      int prow = r + hi * 8;
      Pw[prow * 72 + ln]      = (_Float16)p0;
      Pw[prow * 72 + ln + 16] = (_Float16)p1;
      Pw[prow * 72 + ln + 32] = (_Float16)p2;
      Pw[prow * 72 + ln + 48] = (_Float16)p3;
    }

    // P (16x64) through wave-private LDS into A-fragments (2 k-chunks);
    // batched V B-fragments, then 8 PV WMMAs.
    v16h ap0 = load_afrag(Pw + ln * 72 + hi * 8);
    v16h ap1 = load_afrag(Pw + ln * 72 + 32 + hi * 8);
    v16h vf[8];
#pragma unroll
    for (int j = 0; j < 4; ++j) {
      vf[j * 2 + 0] = load_bfrag(Vt + (j * 16 + ln) * 72 + hi * 16);
      vf[j * 2 + 1] = load_bfrag(Vt + (j * 16 + ln) * 72 + 32 + hi * 16);
    }
#pragma unroll
    for (int j = 0; j < 4; ++j) o[j] = wmma_f16(ap0, vf[j * 2 + 0], o[j]);
#pragma unroll
    for (int j = 0; j < 4; ++j) o[j] = wmma_f16(ap1, vf[j * 2 + 1], o[j]);
  }

  // normalize and emit O as f16 in [b, s, h*64+d] layout
  size_t base0 = ((size_t)b * S_LEN + qbase + hi * 8) * DMODEL + h * HDIM + ln;
#pragma unroll
  for (int r = 0; r < 8; ++r) {
    float inv = 1.0f / lrow[r];
    size_t base = base0 + (size_t)r * DMODEL;
    Oh[base +  0] = (_Float16)(o[0][r] * inv);
    Oh[base + 16] = (_Float16)(o[1][r] * inv);
    Oh[base + 32] = (_Float16)(o[2][r] * inv);
    Oh[base + 48] = (_Float16)(o[3][r] * inv);
  }
}

// ------------------------------------------------------ output projection
// grid = (MROWS/128, DMODEL/64); Wo 64k-chunks async-staged, fp32 out + bias.
__global__ __launch_bounds__(256) void k_out_gemm(
    const _Float16* __restrict__ Oh, const _Float16* __restrict__ Wot,
    const float* __restrict__ bo, float* __restrict__ out) {
  __shared__ _Float16 Wl[2][64 * 72];
  const int wave  = threadIdx.x >> 5;
  const int lane  = threadIdx.x & 31;
  const int ln    = lane & 15;
  const int hi    = lane >> 4;
  const int mrow0 = blockIdx.x * 128 + wave * 16;
  const int nb    = blockIdx.y * 64;

  const int c0 = threadIdx.x, c1 = threadIdx.x + 256;
  const int n0 = c0 >> 3, p0_ = (c0 & 7) * 8;
  const int n1 = c1 >> 3, p1_ = (c1 & 7) * 8;
  auto stage_w = [&](int buf, int kc) {
    async_copy16(&Wl[buf][n0 * 72 + p0_],
                 Wot + (size_t)(nb + n0) * DMODEL + kc + p0_);
    async_copy16(&Wl[buf][n1 * 72 + p1_],
                 Wot + (size_t)(nb + n1) * DMODEL + kc + p1_);
  };

  v8f acc[4] = {{}, {}, {}, {}};
  const _Float16* arowp = Oh + (size_t)(mrow0 + ln) * DMODEL + hi * 8;

  stage_w(0, 0);
  v16h a0 = load_afrag(arowp);
  v16h a1 = load_afrag(arowp + 32);
  for (int kc = 0; kc < DMODEL; kc += 64) {
    const int cur = (kc >> 6) & 1;
    wait_async0();
    __syncthreads();
    if (kc + 64 < DMODEL) stage_w(cur ^ 1, kc + 64);
    v16h ac0 = a0, ac1 = a1;
    if (kc + 64 < DMODEL) {
      a0 = load_afrag(arowp + kc + 64);
      a1 = load_afrag(arowp + kc + 96);
    }
    const _Float16* Wb = &Wl[cur][0];
    v16h bf[8];
#pragma unroll
    for (int j = 0; j < 4; ++j) {
      bf[j * 2 + 0] = load_bfrag(Wb + (j * 16 + ln) * 72 + hi * 16);
      bf[j * 2 + 1] = load_bfrag(Wb + (j * 16 + ln) * 72 + 32 + hi * 16);
    }
#pragma unroll
    for (int j = 0; j < 4; ++j) acc[j] = wmma_f16(ac0, bf[j * 2 + 0], acc[j]);
#pragma unroll
    for (int j = 0; j < 4; ++j) acc[j] = wmma_f16(ac1, bf[j * 2 + 1], acc[j]);
  }
  float b0 = bo[nb + ln], b1 = bo[nb + 16 + ln], b2 = bo[nb + 32 + ln],
        b3 = bo[nb + 48 + ln];
#pragma unroll
  for (int r = 0; r < 8; ++r) {
    float* op = out + (size_t)(mrow0 + r + hi * 8) * DMODEL + nb + ln;
    op[0]  = acc[0][r] + b0;
    op[16] = acc[1][r] + b1;
    op[32] = acc[2][r] + b2;
    op[48] = acc[3][r] + b3;
  }
}

// ---------------------------------------------------------------- launch
extern "C" void kernel_launch(void* const* d_in, const int* in_sizes, int n_in,
                              void* d_out, int out_size, void* d_ws, size_t ws_size,
                              hipStream_t stream) {
  const float* x  = (const float*)d_in[0];
  const float* Wq = (const float*)d_in[1];
  const float* bq = (const float*)d_in[2];
  const float* Wk = (const float*)d_in[3];
  const float* bk = (const float*)d_in[4];
  const float* Wv = (const float*)d_in[5];
  const float* bv = (const float*)d_in[6];
  const float* Wo = (const float*)d_in[7];
  const float* bo = (const float*)d_in[8];
  float* out = (float*)d_out;

  char* ws = (char*)d_ws;
  const size_t MB = 1024ull * 1024ull;
  _Float16* xh  = (_Float16*)(ws +  0 * MB);  // 8 MB (x f16; reused as O f16)
  _Float16* Wqt = (_Float16*)(ws +  8 * MB);  // 2 MB  [n=h*64+d][f]
  _Float16* Wkt = (_Float16*)(ws + 10 * MB);  // 2 MB
  _Float16* Wvt = (_Float16*)(ws + 12 * MB);  // 2 MB
  _Float16* Wot = (_Float16*)(ws + 14 * MB);  // 2 MB  [f][h*64+d]
  _Float16* Qb  = (_Float16*)(ws + 16 * MB);  // 8 MB  [b,h,s,d]
  _Float16* Kb  = (_Float16*)(ws + 24 * MB);  // 8 MB  [b,h,s,d]
  _Float16* Vtb = (_Float16*)(ws + 32 * MB);  // 8 MB  [b,h,d,s]

  const int nx = MROWS * DMODEL;
  k_f32_to_f16<<<(nx + 255) / 256, 256, 0, stream>>>(x, xh, nx);
  const int nw = DMODEL * DMODEL;
  const int gw = (nw + 255) / 256;
  k_transpose_f32_to_f16<<<gw, 256, 0, stream>>>(Wq, Wqt, DMODEL, DMODEL);
  k_transpose_f32_to_f16<<<gw, 256, 0, stream>>>(Wk, Wkt, DMODEL, DMODEL);
  k_transpose_f32_to_f16<<<gw, 256, 0, stream>>>(Wv, Wvt, DMODEL, DMODEL);
  k_transpose_f32_to_f16<<<gw, 256, 0, stream>>>(Wo, Wot, DMODEL, DMODEL);

  k_qkv_gemm<<<dim3(MROWS / 128, NHEADS, 3), 256, 0, stream>>>(
      xh, Wqt, Wkt, Wvt, bq, bk, bv, Qb, Kb, Vtb);

  k_attention<<<dim3(S_LEN / 128, BATCH * NHEADS), 256, 0, stream>>>(
      Qb, Kb, Vtb, /*Oh=*/xh);

  k_out_gemm<<<dim3(MROWS / 128, DMODEL / 64), 256, 0, stream>>>(
      xh, Wot, bo, out);
}
